// EfficientRelativeAttention_51393578664623
// MI455X (gfx1250) — matrix operand order repaired and checked by reading, not compiled, not run on hardware
//
#include <hip/hip_runtime.h>

// ---------------------------------------------------------------------------
// MI455X (gfx1250) relative attention.  wave32, WMMA f16->f32, TDM K-staging.
// ---------------------------------------------------------------------------

typedef __attribute__((ext_vector_type(16))) _Float16 v16h;
typedef __attribute__((ext_vector_type(8)))  _Float16 h8;
typedef __attribute__((ext_vector_type(8)))  float    v8f;
typedef __attribute__((ext_vector_type(4)))  float    f4;
typedef __attribute__((ext_vector_type(4)))  unsigned int u32x4;
typedef __attribute__((ext_vector_type(8)))  int      i32x8;
typedef __attribute__((ext_vector_type(4)))  int      i32x4;

#if __has_builtin(__builtin_amdgcn_tensor_load_to_lds) && \
    __has_builtin(__builtin_amdgcn_s_wait_tensorcnt)
#define HAVE_TDM 1
#else
#define HAVE_TDM 0
#endif

#define WMMA_F16(A, B, C) \
  __builtin_amdgcn_wmma_f32_16x16x32_f16(false, (A), false, (B), (short)0, (C), false, false)

// Build a v16h fragment from two contiguous 16-byte chunks.
static __device__ __forceinline__ v16h frag2(const _Float16* p0, const _Float16* p1) {
  h8 a = *(const h8*)p0;
  h8 b = *(const h8*)p1;
  v16h r;
#pragma unroll
  for (int i = 0; i < 8; ++i) { r[i] = a[i]; r[i + 8] = b[i]; }
  return r;
}

// Reductions across the 16 lanes of a wave32 half (rows of a C-tile half).
static __device__ __forceinline__ float rmax16(float v) {
  v = fmaxf(v, __shfl_xor(v, 1, 32));
  v = fmaxf(v, __shfl_xor(v, 2, 32));
  v = fmaxf(v, __shfl_xor(v, 4, 32));
  v = fmaxf(v, __shfl_xor(v, 8, 32));
  return v;
}
static __device__ __forceinline__ float rsum16(float v) {
  v += __shfl_xor(v, 1, 32);
  v += __shfl_xor(v, 2, 32);
  v += __shfl_xor(v, 4, 32);
  v += __shfl_xor(v, 8, 32);
  return v;
}

// ---------------------------------------------------------------------------
// f32 -> f16 conversion kernels
// ---------------------------------------------------------------------------
__global__ void k_cvt(const float* __restrict__ s, _Float16* __restrict__ d, int n) {
  int i = blockIdx.x * 256 + threadIdx.x;
  if (i < n) d[i] = (_Float16)s[i];
}

// rel_k slice [moff, moff+2047) -> 2048 rows (row 2047 zero padding).
__global__ void k_cvt_rk(const float* __restrict__ s, _Float16* __restrict__ d, int moff) {
  int i = blockIdx.x * 256 + threadIdx.x;
  if (i >= 2048 * 64) return;
  d[i] = (i < 2047 * 64) ? (_Float16)s[moff * 64 + i] : (_Float16)0.0f;
}

// ---------------------------------------------------------------------------
// GEMM core: one wave computes a 32x64 tile of C = A @ W^T, K = 1024.
// Software-pipelined: fragments for k+1 are issued before the 8 WMMAs that
// consume k's registers, so s_wait_loadcnt overlaps with matrix math.
// 12 b128 loads per 8 WMMAs.
// ---------------------------------------------------------------------------
static __device__ __forceinline__ void gemm32x64(const _Float16* __restrict__ A0,
                                                 const _Float16* __restrict__ W0,
                                                 v8f acc[8]) {
  const int lane = threadIdx.x & 31;
  const int n = lane & 15, hh = lane >> 4;
  const _Float16* ar0 = A0 + (size_t)n * 1024;         // rows r0 .. r0+15
  const _Float16* ar1 = A0 + (size_t)(16 + n) * 1024;  // rows r0+16 .. r0+31
  const _Float16* wr0 = W0 + (size_t)n * 1024;
  const _Float16* wr1 = wr0 + (size_t)16 * 1024;
  const _Float16* wr2 = wr0 + (size_t)32 * 1024;
  const _Float16* wr3 = wr0 + (size_t)48 * 1024;
  const int ao = 8 * hh, bo = 16 * hh;

  v16h a0 = frag2(ar0 + bo, ar0 + bo + 8);
  v16h a1 = frag2(ar1 + bo, ar1 + bo + 8);
  v16h b0 = frag2(wr0 + bo, wr0 + bo + 8);
  v16h b1 = frag2(wr1 + bo, wr1 + bo + 8);
  v16h b2 = frag2(wr2 + bo, wr2 + bo + 8);
  v16h b3 = frag2(wr3 + bo, wr3 + bo + 8);

#pragma unroll 2
  for (int k0 = 0; k0 < 1024; k0 += 32) {
    const int k1 = (k0 + 32) & 1023;  // last iter wraps: harmless dead prefetch
    v16h a0n = frag2(ar0 + k1 + bo, ar0 + k1 + bo + 8);
    v16h a1n = frag2(ar1 + k1 + bo, ar1 + k1 + bo + 8);
    v16h b0n = frag2(wr0 + k1 + bo, wr0 + k1 + bo + 8);
    v16h b1n = frag2(wr1 + k1 + bo, wr1 + k1 + bo + 8);
    v16h b2n = frag2(wr2 + k1 + bo, wr2 + k1 + bo + 8);
    v16h b3n = frag2(wr3 + k1 + bo, wr3 + k1 + bo + 8);

    acc[0] = WMMA_F16(a0, b0, acc[0]);
    acc[1] = WMMA_F16(a0, b1, acc[1]);
    acc[2] = WMMA_F16(a0, b2, acc[2]);
    acc[3] = WMMA_F16(a0, b3, acc[3]);
    acc[4] = WMMA_F16(a1, b0, acc[4]);
    acc[5] = WMMA_F16(a1, b1, acc[5]);
    acc[6] = WMMA_F16(a1, b2, acc[6]);
    acc[7] = WMMA_F16(a1, b3, acc[7]);

    a0 = a0n; a1 = a1n; b0 = b0n; b1 = b1n; b2 = b2n; b3 = b3n;
  }
}

// qkv = x @ qkv_w^T, scattered into q (pre-scaled by hd^-0.5), k, v (B,H,L,hd) f16.
__global__ __launch_bounds__(256) void k_gemm_qkv(const _Float16* __restrict__ X,
                                                  const _Float16* __restrict__ W,
                                                  _Float16* __restrict__ qb,
                                                  _Float16* __restrict__ kb,
                                                  _Float16* __restrict__ vb) {
  const int wave = threadIdx.x >> 5, lane = threadIdx.x & 31;
  const int n = lane & 15, hh = lane >> 4;
  const int r0 = blockIdx.x * 256 + wave * 32;   // row in [0,4096)
  const int c0 = blockIdx.y * 64;                // col in [0,3072)
  v8f acc[8] = {};
  gemm32x64(X + (size_t)r0 * 1024, W + (size_t)c0 * 1024, acc);

  const int sec  = c0 >> 10;             // 0=q 1=k 2=v (64-col tile never crosses)
  const int head = (c0 & 1023) >> 6;
  _Float16* dst = (sec == 0) ? qb : ((sec == 1) ? kb : vb);
  const float sc = (sec == 0) ? 0.125f : 1.0f;   // hd^-0.5 folded into q
#pragma unroll
  for (int rt = 0; rt < 2; ++rt)
#pragma unroll
    for (int fb = 0; fb < 4; ++fb)
#pragma unroll
      for (int r = 0; r < 8; ++r) {
        int row = r0 + rt * 16 + r + 8 * hh;     // b*1024 + i
        int b = row >> 10, i = row & 1023;
        dst[(((size_t)(b * 16 + head) * 1024 + i) << 6) + fb * 16 + n] =
            (_Float16)(acc[rt * 4 + fb][r] * sc);
      }
}

// out = attn_out @ out_w^T  (f32 output)
__global__ __launch_bounds__(256) void k_gemm_out(const _Float16* __restrict__ Ain,
                                                  const _Float16* __restrict__ W,
                                                  float* __restrict__ out) {
  const int wave = threadIdx.x >> 5, lane = threadIdx.x & 31;
  const int n = lane & 15, hh = lane >> 4;
  const int r0 = blockIdx.x * 256 + wave * 32;
  const int c0 = blockIdx.y * 64;
  v8f acc[8] = {};
  gemm32x64(Ain + (size_t)r0 * 1024, W + (size_t)c0 * 1024, acc);
#pragma unroll
  for (int rt = 0; rt < 2; ++rt)
#pragma unroll
    for (int fb = 0; fb < 4; ++fb)
#pragma unroll
      for (int r = 0; r < 8; ++r)
        out[(size_t)(r0 + rt * 16 + r + 8 * hh) * 1024 + c0 + fb * 16 + n] =
            acc[rt * 4 + fb][r];
}

// ---------------------------------------------------------------------------
// Flash attention with relative-position skew.
//   scores[i,j] = q_i . k_j + q_i . rk[j - i + L - 1]   (scale folded in q)
// grid = (L/128, B*H); 8 waves/WG, each wave owns one 16-row query tile.
// The shared 32x64 K tile is staged into LDS by the Tensor Data Mover
// (one descriptor per block, TDM pad feature -> 36-dword rows), waited with
// s_wait_tensorcnt.  Rel band q.rk computed as 3 WMMA tiles, skew gathered
// with ds_bpermute.  Online softmax, P transposed via per-wave LDS staging,
// V transposed in LDS for the P@V WMMAs.
// ---------------------------------------------------------------------------
__global__ __launch_bounds__(256) void k_attn(const _Float16* __restrict__ qb,
                                              const _Float16* __restrict__ kb,
                                              const _Float16* __restrict__ vb,
                                              const _Float16* __restrict__ rk,
                                              _Float16* __restrict__ ao) {
  __shared__ _Float16 sVT[64 * 40];      // V transposed: [d][j], stride 40 (16B align)
  __shared__ float    sP[8][16 * 36];    // per-wave P staging, stride 36 (144B rows)
#if HAVE_TDM
  __shared__ _Float16 sK[32 * 72];       // K tile [j][d], TDM-padded rows (72 halfs)
#endif

  const int tid = threadIdx.x;
  const int wave = tid >> 5, lane = tid & 31;
  const int n = lane & 15, hh = lane >> 4;
  const int bh = blockIdx.y, b = bh >> 4, h = bh & 15;
  const int i0 = blockIdx.x * 128 + wave * 16;

  const _Float16* qp = qb + (size_t)bh * 1024 * 64;
  const _Float16* kp = kb + (size_t)bh * 1024 * 64;
  const _Float16* vp = vb + (size_t)bh * 1024 * 64;

  // Q as two A fragments (d 0..31, 32..63), loaded once.
  const _Float16* qrow = qp + (size_t)(i0 + n) * 64;
  const v16h qa0 = frag2(qrow + 16 * hh,      qrow + 16 * hh + 8);
  const v16h qa1 = frag2(qrow + 32 + 16 * hh, qrow + 32 + 16 * hh + 8);

  v8f oa[4] = {};
  float mrow[8], lrow[8];
#pragma unroll
  for (int r = 0; r < 8; ++r) { mrow[r] = -1e30f; lrow[r] = 0.0f; }

  float* sPw = sP[wave];

  for (int j0 = 0; j0 < 1024; j0 += 32) {
    __syncthreads();
#if HAVE_TDM
    if (wave == 0) {  // TDM: DMA the shared K tile (32 rows x 64 f16) into LDS
      const unsigned long long ga = (unsigned long long)(uintptr_t)(kp + (size_t)j0 * 64);
      u32x4 g0;
      g0[0] = 1u;                                        // count=1, normal mode
      g0[1] = (unsigned int)(uintptr_t)(void*)sK;        // lds_addr
      g0[2] = (unsigned int)ga;                          // global_addr[31:0]
      g0[3] = (unsigned int)((ga >> 32) & 0x1FFFFFFu) | (2u << 30);  // [56:32], type=2
      i32x8 g1;
      g1[0] = (1 << 16) | (1 << 20) | (4 << 22) | (3 << 25);
      //       data=2B     pad_en     every 32dw  pad 4dw  -> 36-dword LDS rows
      g1[1] = (64 << 16);   // tensor_dim0 = 64
      g1[2] = (32 << 16);   // tensor_dim1 = 32
      g1[3] = (64 << 16);   // tile_dim0 = 64
      g1[4] = 32;           // tile_dim1 = 32, tile_dim2 = 0
      g1[5] = 64;           // tensor_dim0_stride = 64
      g1[6] = 0; g1[7] = 0;
      i32x4 gz4 = {0, 0, 0, 0};
      i32x8 gz8 = {0, 0, 0, 0, 0, 0, 0, 0};
      __builtin_amdgcn_tensor_load_to_lds(g0, g1, gz4, gz4, gz8, 0);
    }
#endif
    {  // cooperative transposed V load: 32 keys x 64 dims -> sVT[d][j]
      int j = tid >> 3, dg = (tid & 7) * 8;
      h8 vv = *(const h8*)(vp + (size_t)(j0 + j) * 64 + dg);
#pragma unroll
      for (int e = 0; e < 8; ++e) sVT[(dg + e) * 40 + j] = vv[e];
    }
#if HAVE_TDM
    if (wave == 0) __builtin_amdgcn_s_wait_tensorcnt(0);
#endif
    __syncthreads();

    // ---- content scores: two 16x16 tiles over this 32-key block ----
#if HAVE_TDM
    const _Float16* kr0 = sK + (size_t)n * 72;
    const _Float16* kr1 = sK + (size_t)(16 + n) * 72;
#else
    if (j0 + 32 < 1024)
      __builtin_prefetch(kp + (size_t)(j0 + 32 + n) * 64, 0, 3);
    const _Float16* kr0 = kp + (size_t)(j0 + n) * 64;
    const _Float16* kr1 = kp + (size_t)(j0 + 16 + n) * 64;
#endif
    v8f Sa = {}, Sb = {};
    Sa = WMMA_F16(qa0, frag2(kr0 + 16 * hh,      kr0 + 16 * hh + 8),      Sa);
    Sa = WMMA_F16(qa1, frag2(kr0 + 32 + 16 * hh, kr0 + 32 + 16 * hh + 8), Sa);
    Sb = WMMA_F16(qa0, frag2(kr1 + 16 * hh,      kr1 + 16 * hh + 8),      Sb);
    Sb = WMMA_F16(qa1, frag2(kr1 + 32 + 16 * hh, kr1 + 32 + 16 * hh + 8), Sb);

    // ---- rel logits band: P[u,t] = q_u . rk[mbase + t], t = 0..47 ----
    const int mbase = 1023 + j0 - i0 - 15;     // always in [0, 2000]
    v8f P[3];
#pragma unroll
    for (int p = 0; p < 3; ++p) {
      const _Float16* rr = rk + (size_t)(mbase + p * 16 + n) * 64;
      v8f acc = {};
      acc = WMMA_F16(qa0, frag2(rr + 16 * hh,      rr + 16 * hh + 8),      acc);
      acc = WMMA_F16(qa1, frag2(rr + 32 + 16 * hh, rr + 32 + 16 * hh + 8), acc);
      P[p] = acc;
    }

    // ---- Toeplitz skew gather: rel[u,w] = P[15 + w - u] via ds_bpermute ----
#pragma unroll
    for (int r = 0; r < 8; ++r) {
      int t = 15 + n - (r + 8 * hh);                    // in [0,30]
      int idx = ((lane & 16) | (t & 15)) << 2;          // stay within this half
      float ga = __int_as_float(__builtin_amdgcn_ds_bpermute(idx, __float_as_int(P[0][r])));
      float gb = __int_as_float(__builtin_amdgcn_ds_bpermute(idx, __float_as_int(P[1][r])));
      float gc = __int_as_float(__builtin_amdgcn_ds_bpermute(idx, __float_as_int(P[2][r])));
      Sa[r] += (t < 16) ? ga : gb;                      // tile 1: t' = t
      Sb[r] += (t < 16) ? gb : gc;                      // tile 2: t' = t + 16
    }

    // ---- online softmax ----
    float alpha[8];
#pragma unroll
    for (int r = 0; r < 8; ++r) {
      float mb = rmax16(fmaxf(Sa[r], Sb[r]));
      float mn = fmaxf(mrow[r], mb);
      alpha[r] = __expf(mrow[r] - mn);
      mrow[r] = mn;
      float pa = __expf(Sa[r] - mn);
      float pb = __expf(Sb[r] - mn);
      lrow[r] = lrow[r] * alpha[r] + rsum16(pa + pb);
      sPw[(r + 8 * hh) * 36 + n]      = pa;             // stage P for transpose
      sPw[(r + 8 * hh) * 36 + 16 + n] = pb;
    }
#pragma unroll
    for (int db = 0; db < 4; ++db)
#pragma unroll
      for (int r = 0; r < 8; ++r) oa[db][r] *= alpha[r];

    // ---- rebuild P as 16x32 A fragment from staging ----
    const float* prow = sPw + n * 36;
    f4 f0 = *(const f4*)(prow + 16 * hh);
    f4 f1 = *(const f4*)(prow + 16 * hh + 4);
    f4 f2 = *(const f4*)(prow + 16 * hh + 8);
    f4 f3 = *(const f4*)(prow + 16 * hh + 12);
    v16h pf;
#pragma unroll
    for (int e = 0; e < 4; ++e) {
      pf[e]      = (_Float16)f0[e];
      pf[e + 4]  = (_Float16)f1[e];
      pf[e + 8]  = (_Float16)f2[e];
      pf[e + 12] = (_Float16)f3[e];
    }

    // ---- O += P @ V (V fragments from transposed LDS, 16B-contiguous) ----
#pragma unroll
    for (int db = 0; db < 4; ++db) {
      const _Float16* vt = sVT + (db * 16 + n) * 40 + 16 * hh;
      h8 lo = *(const h8*)vt;
      h8 hi = *(const h8*)(vt + 8);
      v16h vf;
#pragma unroll
      for (int e = 0; e < 8; ++e) { vf[e] = lo[e]; vf[e + 8] = hi[e]; }
      oa[db] = WMMA_F16(pf, vf, oa[db]);
    }
  }

  // epilogue: normalize by row sums, store f16 at (b, i, h*64 + d)
#pragma unroll
  for (int db = 0; db < 4; ++db)
#pragma unroll
    for (int r = 0; r < 8; ++r) {
      int i = i0 + r + 8 * hh;
      ao[(size_t)(b * 1024 + i) * 1024 + h * 64 + db * 16 + n] =
          (_Float16)(oa[db][r] / lrow[r]);
    }
}

// ---------------------------------------------------------------------------
// host launch
// ---------------------------------------------------------------------------
extern "C" void kernel_launch(void* const* d_in, const int* in_sizes, int n_in,
                              void* d_out, int out_size, void* d_ws, size_t ws_size,
                              hipStream_t stream) {
  const float* x     = (const float*)d_in[0];
  const float* qkv_w = (const float*)d_in[1];
  const float* out_w = (const float*)d_in[2];
  const float* rel_k = (const float*)d_in[3];

  const int rel_rows = in_sizes[3] / 64;       // 2*max_len - 1
  const int max_len  = (rel_rows + 1) / 2;
  const int moff     = max_len - 1024;         // slice start: max_len - L

  char* w = (char*)d_ws;
  auto alloc = [&](size_t bytes) -> char* {
    char* p = w; w += (bytes + 255) & ~(size_t)255; return p;
  };
  _Float16* xbf  = (_Float16*)alloc((size_t)4096 * 1024 * 2);
  _Float16* wq   = (_Float16*)alloc((size_t)3072 * 1024 * 2);
  _Float16* wo   = (_Float16*)alloc((size_t)1024 * 1024 * 2);
  _Float16* rkbf = (_Float16*)alloc((size_t)2048 * 64 * 2);
  _Float16* qbf  = (_Float16*)alloc((size_t)4 * 16 * 1024 * 64 * 2);
  _Float16* kbf  = (_Float16*)alloc((size_t)4 * 16 * 1024 * 64 * 2);
  _Float16* vbf  = (_Float16*)alloc((size_t)4 * 16 * 1024 * 64 * 2);
  _Float16* abf  = (_Float16*)alloc((size_t)4096 * 1024 * 2);

  k_cvt<<<(4096 * 1024 + 255) / 256, 256, 0, stream>>>(x, xbf, 4096 * 1024);
  k_cvt<<<(3072 * 1024 + 255) / 256, 256, 0, stream>>>(qkv_w, wq, 3072 * 1024);
  k_cvt<<<(1024 * 1024 + 255) / 256, 256, 0, stream>>>(out_w, wo, 1024 * 1024);
  k_cvt_rk<<<(2048 * 64 + 255) / 256, 256, 0, stream>>>(rel_k, rkbf, moff);

  k_gemm_qkv<<<dim3(16, 48), 256, 0, stream>>>(xbf, wq, qbf, kbf, vbf);
  k_attn<<<dim3(8, 64), 256, 0, stream>>>(qbf, kbf, vbf, rkbf, abf);
  k_gemm_out<<<dim3(16, 16), 256, 0, stream>>>(abf, wo, (float*)d_out);
}
